// GPT2ParallelSelfAttention_27401891348587
// MI455X (gfx1250) — matrix operand order repaired and checked by reading, not compiled, not run on hardware
//
#include <hip/hip_runtime.h>
#include <hip/hip_bf16.h>

typedef __attribute__((ext_vector_type(16))) _Float16 v16h;
typedef __attribute__((ext_vector_type(8)))  _Float16 v8h;
typedef __attribute__((ext_vector_type(8)))  float    v8f;

#define B_   4
#define TEXT_ 64
#define S0_  1088
#define S1_  4096
#define S_   5184
#define H_   1024
#define NH_  16
#define HD_  64

// ---------------- helpers ----------------

__device__ inline v8f zero8f() {
  v8f z = {0.f,0.f,0.f,0.f,0.f,0.f,0.f,0.f};
  return z;
}
__device__ inline v8h zero8h() {
  v8h z = {(_Float16)0,(_Float16)0,(_Float16)0,(_Float16)0,
           (_Float16)0,(_Float16)0,(_Float16)0,(_Float16)0};
  return z;
}
__device__ inline v16h zero16h() {
  v16h z = {(_Float16)0,(_Float16)0,(_Float16)0,(_Float16)0,
            (_Float16)0,(_Float16)0,(_Float16)0,(_Float16)0,
            (_Float16)0,(_Float16)0,(_Float16)0,(_Float16)0,
            (_Float16)0,(_Float16)0,(_Float16)0,(_Float16)0};
  return z;
}

__device__ inline v8f wmma16(v16h a, v16h b, v8f c) {
  return __builtin_amdgcn_wmma_f32_16x16x32_f16(false, a, false, b, (short)0, c, false, false);
}

// A-fragment (16x32 f16): lane half g, row = lane&15.
// element e<8 : K = g*8 + e ; element e>=8 : K = 16 + g*8 + (e-8)
__device__ inline v16h load_a_f16(const _Float16* __restrict__ p /*row base + k0*/, int g) {
  v8h lo = *(const v8h*)(p + g*8);
  v8h hi = *(const v8h*)(p + 16 + g*8);
  return __builtin_shufflevector(lo, hi, 0,1,2,3,4,5,6,7,8,9,10,11,12,13,14,15);
}
// B-fragment (32x16 f16): lane half g, col = lane&15.
// element e<8 : K = g*8 + e ; element e>=8 : K = 16 + g*8 + (e-8)
__device__ inline v16h load_b_f16(const _Float16* __restrict__ p /*col base + k0*/, int g) {
  v8h lo = *(const v8h*)(p + g*8);
  v8h hi = *(const v8h*)(p + 16 + g*8);
  return __builtin_shufflevector(lo, hi, 0,1,2,3,4,5,6,7,8,9,10,11,12,13,14,15);
}
__device__ inline v16h pack_a_from_f32(const float* __restrict__ p, int g, float scale) {
  v16h r;
  #pragma unroll
  for (int e = 0; e < 8; ++e) r[e]   = (_Float16)(p[g*8 + e] * scale);
  #pragma unroll
  for (int e = 0; e < 8; ++e) r[e+8] = (_Float16)(p[16 + g*8 + e] * scale);
  return r;
}
__device__ inline v16h pack_b_from_f32(const float* __restrict__ p, int g) {
  v16h r;
  #pragma unroll
  for (int e = 0; e < 16; ++e) r[e] = (_Float16)p[(e < 8) ? (g*8 + e) : (16 + g*8 + (e - 8))];
  return r;
}

// ---------------- pack kernels ----------------

__global__ void cvt_f32_f16_kernel(const float* __restrict__ src, _Float16* __restrict__ dst, size_t n) {
  size_t i = (size_t)blockIdx.x * blockDim.x + threadIdx.x;
  if (i < n) dst[i] = (_Float16)src[i];
}

// dst[n*K + k] = src[k*N + n]   (weights K x N -> packed N x K)
__global__ void transpose_cvt_kernel(const float* __restrict__ src, _Float16* __restrict__ dst,
                                     int K, int N) {
  size_t i = (size_t)blockIdx.x * blockDim.x + threadIdx.x;
  if (i >= (size_t)K * N) return;
  int k = (int)(i % K);
  int n = (int)(i / K);
  dst[i] = (_Float16)src[(size_t)k * N + n];
}

// Vp[((b*NH+h)*HD + d)*S0 + s] = qkv[(b*S + s)*3H + 2H + h*HD + d]
__global__ void pack_v0_kernel(const float* __restrict__ qkv, _Float16* __restrict__ Vp) {
  size_t i = (size_t)blockIdx.x * blockDim.x + threadIdx.x;
  if (i >= (size_t)B_ * NH_ * HD_ * S0_) return;
  int s = (int)(i % S0_);
  size_t r = i / S0_;
  int d = (int)(r % HD_); r /= HD_;
  int h = (int)(r % NH_);
  int b = (int)(r / NH_);
  Vp[i] = (_Float16)qkv[(size_t)(b * S_ + s) * (3*H_) + 2*H_ + h*HD_ + d];
}

// ---------------- QKV GEMM (WMMA, software-pipelined 32x64 tile) ----------------

__global__ void __launch_bounds__(256) qkv_gemm_kernel(
    const _Float16* __restrict__ hsb,
    const _Float16* __restrict__ Wq,  const _Float16* __restrict__ Wqp,
    const float* __restrict__ bq, const float* __restrict__ bqp,
    float* __restrict__ qkv)
{
  int wave = (int)((blockIdx.x * blockDim.x + threadIdx.x) >> 5);
  int lane = (int)(threadIdx.x & 31);
  const int MB = (B_*S_)/32;       // 648
  const int NB = (3*H_)/64;        // 48
  int mb = wave / NB;
  int nb = wave % NB;
  if (mb >= MB) return;
  int m0 = mb * 32;
  int n0 = nb * 64;
  int b  = m0 / S_;
  int sl = m0 - b * S_;
  const _Float16* W = (sl < S0_) ? Wq : Wqp;
  const float* bias = (sl < S0_) ? bq : bqp;

  int g = lane >> 4, li = lane & 15;
  const _Float16* arow0 = hsb + (size_t)(m0 + li) * H_;
  const _Float16* arow1 = arow0 + (size_t)16 * H_;
  const _Float16* brow0 = W + (size_t)(n0 +  0 + li) * H_;
  const _Float16* brow1 = W + (size_t)(n0 + 16 + li) * H_;
  const _Float16* brow2 = W + (size_t)(n0 + 32 + li) * H_;
  const _Float16* brow3 = W + (size_t)(n0 + 48 + li) * H_;

  v8f acc[8];
  #pragma unroll
  for (int t = 0; t < 8; ++t) acc[t] = zero8f();

  // prologue: load K-step 0
  v16h a0 = load_a_f16(arow0, g);
  v16h a1 = load_a_f16(arow1, g);
  v16h b0 = load_b_f16(brow0, g);
  v16h b1 = load_b_f16(brow1, g);
  v16h b2 = load_b_f16(brow2, g);
  v16h b3 = load_b_f16(brow3, g);

  for (int k0 = 32; k0 < H_; k0 += 32) {
    // issue next K-step's loads before consuming current fragments
    v16h a0n = load_a_f16(arow0 + k0, g);
    v16h a1n = load_a_f16(arow1 + k0, g);
    v16h b0n = load_b_f16(brow0 + k0, g);
    v16h b1n = load_b_f16(brow1 + k0, g);
    v16h b2n = load_b_f16(brow2 + k0, g);
    v16h b3n = load_b_f16(brow3 + k0, g);

    acc[0] = wmma16(a0, b0, acc[0]);
    acc[1] = wmma16(a0, b1, acc[1]);
    acc[2] = wmma16(a0, b2, acc[2]);
    acc[3] = wmma16(a0, b3, acc[3]);
    acc[4] = wmma16(a1, b0, acc[4]);
    acc[5] = wmma16(a1, b1, acc[5]);
    acc[6] = wmma16(a1, b2, acc[6]);
    acc[7] = wmma16(a1, b3, acc[7]);

    a0 = a0n; a1 = a1n; b0 = b0n; b1 = b1n; b2 = b2n; b3 = b3n;
  }
  // epilogue: last K-step
  acc[0] = wmma16(a0, b0, acc[0]);
  acc[1] = wmma16(a0, b1, acc[1]);
  acc[2] = wmma16(a0, b2, acc[2]);
  acc[3] = wmma16(a0, b3, acc[3]);
  acc[4] = wmma16(a1, b0, acc[4]);
  acc[5] = wmma16(a1, b1, acc[5]);
  acc[6] = wmma16(a1, b2, acc[6]);
  acc[7] = wmma16(a1, b3, acc[7]);

  #pragma unroll
  for (int mt = 0; mt < 2; ++mt) {
    #pragma unroll
    for (int t = 0; t < 4; ++t) {
      int n = n0 + t*16 + li;
      float bb = bias[n];
      v8f a = acc[mt*4 + t];
      #pragma unroll
      for (int v = 0; v < 8; ++v) {
        int row = m0 + mt*16 + v + 8*g;
        qkv[(size_t)row * (3*H_) + n] = a[v] + bb;
      }
    }
  }
}

// ---------------- dense causal attention (flash, WMMA) ----------------
// one wave per (b, h, 16-row tile); writes ctx f16 rows [b*S, b*S+S0)

__global__ void __launch_bounds__(32) attn0_kernel(
    const float* __restrict__ qkv, const _Float16* __restrict__ Vp,
    _Float16* __restrict__ ctx)
{
  __shared__ __align__(16) _Float16 pt[16][16];
  int lane = (int)(threadIdx.x & 31);
  int t = (int)blockIdx.x;
  const int NT = S0_/16;  // 68
  int it = t % NT; t /= NT;
  int h = t % NH_;
  int b = t / NH_;
  int g = lane >> 4, li = lane & 15;
  int gr0 = b * S_ + it*16;
  int bh = b * NH_ + h;

  const float* qrow = qkv + (size_t)(gr0 + li) * (3*H_) + h*HD_;
  v16h qa0 = pack_a_from_f32(qrow +  0, g, 0.125f);
  v16h qa1 = pack_a_from_f32(qrow + 32, g, 0.125f);

  float mrow[8], lrow[8];
  v8f o[4];
  #pragma unroll
  for (int v = 0; v < 8; ++v) { mrow[v] = -1e30f; lrow[v] = 0.f; }
  #pragma unroll
  for (int c = 0; c < 4; ++c) o[c] = zero8f();

  for (int jt = 0; jt <= it; ++jt) {
    const float* krow = qkv + (size_t)(b*S_ + jt*16 + li) * (3*H_) + H_ + h*HD_;
    v16h kb0 = pack_b_from_f32(krow +  0, g);
    v16h kb1 = pack_b_from_f32(krow + 32, g);
    v8f s = zero8f();
    s = wmma16(qa0, kb0, s);
    s = wmma16(qa1, kb1, s);

    if (jt == it) {
      #pragma unroll
      for (int v = 0; v < 8; ++v) {
        int rr = v + 8*g;
        if (li > rr) s[v] = -10000.f;
      }
    }
    float mnew[8], f[8], p_[8];
    #pragma unroll
    for (int v = 0; v < 8; ++v) {
      float x = s[v];
      x = fmaxf(x, __shfl_xor(x, 1, 16));
      x = fmaxf(x, __shfl_xor(x, 2, 16));
      x = fmaxf(x, __shfl_xor(x, 4, 16));
      x = fmaxf(x, __shfl_xor(x, 8, 16));
      mnew[v] = fmaxf(mrow[v], x);
      f[v] = __expf(mrow[v] - mnew[v]);
      mrow[v] = mnew[v];
    }
    #pragma unroll
    for (int v = 0; v < 8; ++v) {
      float pv = __expf(s[v] - mnew[v]);
      p_[v] = pv;
      float rs = pv;
      rs += __shfl_xor(rs, 1, 16);
      rs += __shfl_xor(rs, 2, 16);
      rs += __shfl_xor(rs, 4, 16);
      rs += __shfl_xor(rs, 8, 16);
      lrow[v] = lrow[v] * f[v] + rs;
    }
    #pragma unroll
    for (int c = 0; c < 4; ++c)
      #pragma unroll
      for (int v = 0; v < 8; ++v) o[c][v] *= f[v];

    // transpose P (S-layout -> A-layout) through LDS
    __syncthreads();
    #pragma unroll
    for (int v = 0; v < 8; ++v) pt[v + 8*g][li] = (_Float16)p_[v];
    __syncthreads();
    v8h plo = *(const v8h*)&pt[li][g*8];
    v16h pa = __builtin_shufflevector(plo, zero8h(), 0,1,2,3,4,5,6,7,8,9,10,11,12,13,14,15);

    #pragma unroll
    for (int c = 0; c < 4; ++c) {
      v16h vb;
      {
        const _Float16* vp = Vp + ((size_t)(bh*HD_ + c*16 + li)) * S0_ + jt*16 + g*8;
        v8h v0 = *(const v8h*)vp;
        vb = __builtin_shufflevector(v0, zero8h(), 0,1,2,3,4,5,6,7,8,9,10,11,12,13,14,15);
      }
      o[c] = wmma16(pa, vb, o[c]);
    }
  }

  #pragma unroll
  for (int v = 0; v < 8; ++v) {
    float inv = 1.0f / lrow[v];
    int row = gr0 + v + 8*g;
    #pragma unroll
    for (int c = 0; c < 4; ++c) {
      ctx[(size_t)row * H_ + h*HD_ + c*16 + li] = (_Float16)(o[c][v] * inv);
    }
  }
}

// ---------------- local (causal-window + cross) attention ----------------
// one thread per (b, h, grid position); two-pass softmax with recompute

__global__ void __launch_bounds__(256) attn1_kernel(
    const float* __restrict__ qkv, _Float16* __restrict__ ctx)
{
  int t = (int)(blockIdx.x * blockDim.x + threadIdx.x);
  if (t >= B_*NH_*S1_) return;
  int pos = t & (S1_ - 1);
  int bh = t >> 12;
  int h = bh % NH_, b = bh / NH_;
  int y = pos >> 6, x = pos & 63;
  const size_t ld = 3*H_;

  const float* qp = qkv + (size_t)(b*S_ + S0_ + pos) * ld + h*HD_;
  float q[HD_];
  #pragma unroll
  for (int c = 0; c < HD_; c += 4) {
    float4 f = *(const float4*)(qp + c);
    q[c] = f.x*0.125f; q[c+1] = f.y*0.125f; q[c+2] = f.z*0.125f; q[c+3] = f.w*0.125f;
  }
  int ciy = y >> 1, cix = x >> 1;

  // ---- pass 1: max (OOB offsets contribute score 0, like zero-padding) ----
  float m = -1e30f;
  for (int o = 0; o < 49; ++o) {
    int dy = o / 7 - 3, dx = o % 7 - 3;
    int ky = ciy + dy, kx = cix + dx;
    float s = 0.f;
    if (ky >= 0 && ky < 32 && kx >= 0 && kx < 32) {
      const float* kp = qkv + (size_t)(b*S_ + TEXT_ + ky*32 + kx) * ld + H_ + h*HD_;
      #pragma unroll
      for (int c = 0; c < HD_; c += 4) {
        float4 f = *(const float4*)(kp + c);
        s += q[c]*f.x + q[c+1]*f.y + q[c+2]*f.z + q[c+3]*f.w;
      }
    }
    m = fmaxf(m, s);
  }
  for (int o = 0; o < 77; ++o) {
    int dy = o / 9 - 8, dx = o % 9 - 4;
    int ky = y + dy, kx = x + dx;
    float s = 0.f;
    if (ky >= 0 && ky < 64 && kx >= 0 && kx < 64) {
      const float* kp = qkv + (size_t)(b*S_ + S0_ + ky*64 + kx) * ld + H_ + h*HD_;
      #pragma unroll
      for (int c = 0; c < HD_; c += 4) {
        float4 f = *(const float4*)(kp + c);
        s += q[c]*f.x + q[c+1]*f.y + q[c+2]*f.z + q[c+3]*f.w;
      }
    }
    m = fmaxf(m, s);
  }

  // ---- pass 2: exp-sum + weighted accumulate ----
  float acc[HD_];
  #pragma unroll
  for (int c = 0; c < HD_; ++c) acc[c] = 0.f;
  float l = 0.f;

  for (int o = 0; o < 49; ++o) {
    int dy = o / 7 - 3, dx = o % 7 - 3;
    int ky = ciy + dy, kx = cix + dx;
    bool inb = (ky >= 0 && ky < 32 && kx >= 0 && kx < 32);
    float s = 0.f;
    if (inb) {
      const float* kp = qkv + (size_t)(b*S_ + TEXT_ + ky*32 + kx) * ld + H_ + h*HD_;
      #pragma unroll
      for (int c = 0; c < HD_; c += 4) {
        float4 f = *(const float4*)(kp + c);
        s += q[c]*f.x + q[c+1]*f.y + q[c+2]*f.z + q[c+3]*f.w;
      }
    }
    float p = __expf(s - m);
    l += p;
    if (inb) {
      const float* vp = qkv + (size_t)(b*S_ + TEXT_ + ky*32 + kx) * ld + 2*H_ + h*HD_;
      #pragma unroll
      for (int c = 0; c < HD_; c += 4) {
        float4 f = *(const float4*)(vp + c);
        acc[c] += p*f.x; acc[c+1] += p*f.y; acc[c+2] += p*f.z; acc[c+3] += p*f.w;
      }
    }
  }
  for (int o = 0; o < 77; ++o) {
    int dy = o / 9 - 8, dx = o % 9 - 4;
    int ky = y + dy, kx = x + dx;
    bool inb = (ky >= 0 && ky < 64 && kx >= 0 && kx < 64);
    float s = 0.f;
    if (inb) {
      const float* kp = qkv + (size_t)(b*S_ + S0_ + ky*64 + kx) * ld + H_ + h*HD_;
      #pragma unroll
      for (int c = 0; c < HD_; c += 4) {
        float4 f = *(const float4*)(kp + c);
        s += q[c]*f.x + q[c+1]*f.y + q[c+2]*f.z + q[c+3]*f.w;
      }
    }
    float p = __expf(s - m);
    l += p;
    if (inb) {
      const float* vp = qkv + (size_t)(b*S_ + S0_ + ky*64 + kx) * ld + 2*H_ + h*HD_;
      #pragma unroll
      for (int c = 0; c < HD_; c += 4) {
        float4 f = *(const float4*)(vp + c);
        acc[c] += p*f.x; acc[c+1] += p*f.y; acc[c+2] += p*f.z; acc[c+3] += p*f.w;
      }
    }
  }
  float inv = 1.0f / l;
  _Float16* op = ctx + (size_t)(b*S_ + S0_ + pos) * H_ + h*HD_;
  #pragma unroll
  for (int c = 0; c < HD_; ++c) op[c] = (_Float16)(acc[c] * inv);
}

// ---------------- output GEMM (WMMA, software-pipelined 32x64 tile) ----------------

__global__ void __launch_bounds__(256) out_gemm_kernel(
    const _Float16* __restrict__ ctx, const _Float16* __restrict__ Wd,
    const float* __restrict__ bd, float* __restrict__ out)
{
  int wave = (int)((blockIdx.x * blockDim.x + threadIdx.x) >> 5);
  int lane = (int)(threadIdx.x & 31);
  const int MB = (B_*S_)/32;   // 648
  const int NB = H_/64;        // 16
  int mb = wave / NB;
  int nb = wave % NB;
  if (mb >= MB) return;
  int m0 = mb * 32;
  int n0 = nb * 64;
  int g = lane >> 4, li = lane & 15;
  const _Float16* arow0 = ctx + (size_t)(m0 + li) * H_;
  const _Float16* arow1 = arow0 + (size_t)16 * H_;
  const _Float16* brow0 = Wd + (size_t)(n0 +  0 + li) * H_;
  const _Float16* brow1 = Wd + (size_t)(n0 + 16 + li) * H_;
  const _Float16* brow2 = Wd + (size_t)(n0 + 32 + li) * H_;
  const _Float16* brow3 = Wd + (size_t)(n0 + 48 + li) * H_;

  v8f acc[8];
  #pragma unroll
  for (int t = 0; t < 8; ++t) acc[t] = zero8f();

  v16h a0 = load_a_f16(arow0, g);
  v16h a1 = load_a_f16(arow1, g);
  v16h b0 = load_b_f16(brow0, g);
  v16h b1 = load_b_f16(brow1, g);
  v16h b2 = load_b_f16(brow2, g);
  v16h b3 = load_b_f16(brow3, g);

  for (int k0 = 32; k0 < H_; k0 += 32) {
    v16h a0n = load_a_f16(arow0 + k0, g);
    v16h a1n = load_a_f16(arow1 + k0, g);
    v16h b0n = load_b_f16(brow0 + k0, g);
    v16h b1n = load_b_f16(brow1 + k0, g);
    v16h b2n = load_b_f16(brow2 + k0, g);
    v16h b3n = load_b_f16(brow3 + k0, g);

    acc[0] = wmma16(a0, b0, acc[0]);
    acc[1] = wmma16(a0, b1, acc[1]);
    acc[2] = wmma16(a0, b2, acc[2]);
    acc[3] = wmma16(a0, b3, acc[3]);
    acc[4] = wmma16(a1, b0, acc[4]);
    acc[5] = wmma16(a1, b1, acc[5]);
    acc[6] = wmma16(a1, b2, acc[6]);
    acc[7] = wmma16(a1, b3, acc[7]);

    a0 = a0n; a1 = a1n; b0 = b0n; b1 = b1n; b2 = b2n; b3 = b3n;
  }
  acc[0] = wmma16(a0, b0, acc[0]);
  acc[1] = wmma16(a0, b1, acc[1]);
  acc[2] = wmma16(a0, b2, acc[2]);
  acc[3] = wmma16(a0, b3, acc[3]);
  acc[4] = wmma16(a1, b0, acc[4]);
  acc[5] = wmma16(a1, b1, acc[5]);
  acc[6] = wmma16(a1, b2, acc[6]);
  acc[7] = wmma16(a1, b3, acc[7]);

  #pragma unroll
  for (int mt = 0; mt < 2; ++mt) {
    #pragma unroll
    for (int t = 0; t < 4; ++t) {
      int n = n0 + t*16 + li;
      float bb = bd[n];
      v8f a = acc[mt*4 + t];
      #pragma unroll
      for (int v = 0; v < 8; ++v) {
        int row = m0 + mt*16 + v + 8*g;
        out[(size_t)row * H_ + n] = a[v] + bb;
      }
    }
  }
}

// ---------------- host launcher ----------------

extern "C" void kernel_launch(void* const* d_in, const int* in_sizes, int n_in,
                              void* d_out, int out_size, void* d_ws, size_t ws_size,
                              hipStream_t stream) {
  (void)in_sizes; (void)n_in; (void)out_size; (void)ws_size;
  const float* hidden  = (const float*)d_in[0];
  // d_in[1] = mask (unused; causal mask applied analytically)
  const float* W_qkv   = (const float*)d_in[2];
  const float* b_qkv   = (const float*)d_in[3];
  const float* W_qkvp  = (const float*)d_in[4];
  const float* b_qkvp  = (const float*)d_in[5];
  const float* W_dense = (const float*)d_in[6];
  const float* b_dense = (const float*)d_in[7];
  float* out = (float*)d_out;

  char* ws = (char*)d_ws;
  size_t off = 0;
  _Float16* hsb  = (_Float16*)(ws + off); off += (size_t)B_*S_*H_*2;        // f16 activations
  _Float16* Wq_p = (_Float16*)(ws + off); off += (size_t)3*H_*H_*2;         // W_qkv packed [n][k]
  _Float16* Wqp_p= (_Float16*)(ws + off); off += (size_t)3*H_*H_*2;         // W_qkv_plus packed
  _Float16* Wd_p = (_Float16*)(ws + off); off += (size_t)H_*H_*2;           // W_dense packed
  float*    qkv  = (float*)   (ws + off); off += (size_t)B_*S_*3*H_*4;      // QKV f32
  _Float16* ctx  = (_Float16*)(ws + off); off += (size_t)B_*S_*H_*2;        // context f16
  _Float16* Vp   = (_Float16*)(ws + off); off += (size_t)B_*NH_*HD_*S0_*2;  // V0 transposed f16

  // 1) convert activations to f16
  {
    size_t n = (size_t)B_*S_*H_;
    hipLaunchKernelGGL(cvt_f32_f16_kernel, dim3((unsigned)((n + 255)/256)), dim3(256), 0, stream, hidden, hsb, n);
  }
  // 2) pack weights transposed (N-major, contiguous K)
  hipLaunchKernelGGL(transpose_cvt_kernel, dim3((unsigned)(((size_t)H_*3*H_ + 255)/256)), dim3(256), 0, stream,
                     W_qkv, Wq_p, H_, 3*H_);
  hipLaunchKernelGGL(transpose_cvt_kernel, dim3((unsigned)(((size_t)H_*3*H_ + 255)/256)), dim3(256), 0, stream,
                     W_qkvp, Wqp_p, H_, 3*H_);
  hipLaunchKernelGGL(transpose_cvt_kernel, dim3((unsigned)(((size_t)H_*H_ + 255)/256)), dim3(256), 0, stream,
                     W_dense, Wd_p, H_, H_);
  // 3) QKV GEMM (WMMA): 648 * 48 waves, 8 waves/block
  {
    int waves = ((B_*S_)/32) * ((3*H_)/64);
    hipLaunchKernelGGL(qkv_gemm_kernel, dim3((waves + 7)/8), dim3(256), 0, stream,
                       hsb, Wq_p, Wqp_p, b_qkv, b_qkvp, qkv);
  }
  // 4) pack V0 (dense-attention V, [bh][d][token] f16)
  {
    size_t n = (size_t)B_*NH_*HD_*S0_;
    hipLaunchKernelGGL(pack_v0_kernel, dim3((unsigned)((n + 255)/256)), dim3(256), 0, stream, qkv, Vp);
  }
  // 5) dense causal attention: one wave per (b,h,row-tile)
  hipLaunchKernelGGL(attn0_kernel, dim3(B_*NH_*(S0_/16)), dim3(32), 0, stream, qkv, Vp, ctx);
  // 6) local/cross attention: one thread per (b,h,pos)
  hipLaunchKernelGGL(attn1_kernel, dim3((B_*NH_*S1_)/256), dim3(256), 0, stream, qkv, ctx);
  // 7) output GEMM (WMMA): 648 * 16 waves
  {
    int waves = ((B_*S_)/32) * (H_/64);
    hipLaunchKernelGGL(out_gemm_kernel, dim3((waves + 7)/8), dim3(256), 0, stream,
                       ctx, Wd_p, b_dense, out);
  }
}
